// EncoderBlock_41798621725347
// MI455X (gfx1250) — matrix operand order repaired and checked by reading, not compiled, not run on hardware
//
#include <hip/hip_runtime.h>

// ============================================================================
// EncoderBlock for MI455X (gfx1250, wave32, WMMA).
// - All GEMM operands f16 (convert-once); f32 accumulate.
// - GEMM: 128x128x32 tile, 8 waves x (32x64), LDS double-buffered via CDNA5
//   GLOBAL_LOAD_ASYNC_TO_LDS_B128 (+ s_wait_asynccnt); templated epilogue.
// - Attention fused per 16-row q-tile; scores staged f16 in LDS and
//   exponentiated in place so the PV pass is pure ds_load_b128 + WMMA.
// Workspace: ~151 MB.
// ============================================================================

typedef __attribute__((ext_vector_type(16))) _Float16 v16h;
typedef __attribute__((ext_vector_type(8)))  _Float16 v8h;
typedef __attribute__((ext_vector_type(4)))  _Float16 v4h;
typedef __attribute__((ext_vector_type(8)))  float    v8f;
typedef __attribute__((ext_vector_type(4)))  float    v4f;
typedef __attribute__((ext_vector_type(4)))  int      v4i_;

#define B_DIM   8
#define S_LEN   1024
#define D_DIM   512
#define H_DIM   8
#define DH_DIM  64
#define DFF_DIM 2048
#define KW      7
#define NR      33

#if defined(__has_builtin)
#if __has_builtin(__builtin_amdgcn_global_load_async_to_lds_b128) && \
    __has_builtin(__builtin_amdgcn_s_wait_asynccnt)
#define USE_ASYNC_LDS 1
#endif
#endif

#if defined(USE_ASYNC_LDS)
typedef __attribute__((address_space(1))) v4i_ GlobV4;
typedef __attribute__((address_space(3))) v4i_ LdsV4;
__device__ __forceinline__ void async_cp16(const void* g, void* l) {
    __builtin_amdgcn_global_load_async_to_lds_b128((GlobV4*)g, (LdsV4*)l, 0, 0);
}
#endif

// ----------------------------------------------------------------------------
// f16 WMMA GEMM: C = act(A[M,K]h @ B[K,N]h + bias) (+res).
// Compile-time epilogue flags -> straight-line stores.
// ----------------------------------------------------------------------------
#define BM 128
#define BN 128
#define BK 32
#define LDA 40    // padded LDS row stride (halves) for A tile
#define LDB 136   // padded LDS row stride (halves) for B tile

template <int RELU, bool HAS_RES, bool HAS_CF, bool HAS_CH>
__global__ __launch_bounds__(256)
void wmma_gemm_f16(const _Float16* __restrict__ A, const _Float16* __restrict__ Bm,
                   const float* __restrict__ bias, const float* __restrict__ res,
                   float* __restrict__ Cf, _Float16* __restrict__ Ch,
                   int M, int N, int Kd)
{
    __shared__ _Float16 Asm[2][BM * LDA];   // 20.0 KB
    __shared__ _Float16 Bsm[2][BK * LDB];   // 17.0 KB

    const int lane = threadIdx.x;
    const int wave = threadIdx.y;
    const int tid  = wave * 32 + lane;
    const int hf = lane >> 4, rw = lane & 15;
    const int wm = wave & 3, wn = wave >> 2;          // wave: rows wm*32, cols wn*64
    const int rowBase = blockIdx.x * BM;
    const int colBase = blockIdx.y * BN;

    int aRow[2], aCol[2], bRow[2], bCol[2];
    #pragma unroll
    for (int u = 0; u < 2; ++u) {
        int c = tid * 2 + u;
        aRow[u] = c >> 2;  aCol[u] = (c & 3) * 8;     // 128 rows x 4 segs
        bRow[u] = c >> 4;  bCol[u] = (c & 15) * 8;    // 32 rows x 16 segs
    }

    v8f acc[2][4];
    #pragma unroll
    for (int mi = 0; mi < 2; ++mi)
        #pragma unroll
        for (int ni = 0; ni < 4; ++ni)
            #pragma unroll
            for (int e = 0; e < 8; ++e) acc[mi][ni][e] = 0.f;

    auto compute = [&](int b_) {
        const _Float16* As = &Asm[b_][0];
        const _Float16* Bs = &Bsm[b_][0];
        v16h af[2], bf[4];
        #pragma unroll
        for (int mi = 0; mi < 2; ++mi) {
            const _Float16* ap = As + (wm * 32 + mi * 16 + rw) * LDA;
            v8h lo = *(const v8h*)(ap + hf * 8);
            v8h hi = *(const v8h*)(ap + 16 + hf * 8);
            #pragma unroll
            for (int j = 0; j < 8; ++j) { af[mi][j] = lo[j]; af[mi][8 + j] = hi[j]; }
        }
        #pragma unroll
        for (int ni = 0; ni < 4; ++ni) {
            const _Float16* bp = Bs + wn * 64 + ni * 16 + rw;
            #pragma unroll
            for (int j = 0; j < 8; ++j) { bf[ni][j] = bp[(hf * 8 + j) * LDB]; bf[ni][8 + j] = bp[(16 + hf * 8 + j) * LDB]; }
        }
        #pragma unroll
        for (int mi = 0; mi < 2; ++mi)
            #pragma unroll
            for (int ni = 0; ni < 4; ++ni)
                acc[mi][ni] = __builtin_amdgcn_wmma_f32_16x16x32_f16(
                    false, af[mi], false, bf[ni], (short)0, acc[mi][ni], false, false);
    };

    // ---- preload tile 0 into buffer 0 ----
#if defined(USE_ASYNC_LDS)
    #pragma unroll
    for (int u = 0; u < 2; ++u) {
        async_cp16(A  + (size_t)(rowBase + aRow[u]) * Kd + aCol[u],
                   &Asm[0][aRow[u] * LDA + aCol[u]]);
        async_cp16(Bm + (size_t)bRow[u] * N + colBase + bCol[u],
                   &Bsm[0][bRow[u] * LDB + bCol[u]]);
    }
    __builtin_amdgcn_s_wait_asynccnt(0);
#else
    #pragma unroll
    for (int u = 0; u < 2; ++u) {
        v8h va = *(const v8h*)(A  + (size_t)(rowBase + aRow[u]) * Kd + aCol[u]);
        v8h vb = *(const v8h*)(Bm + (size_t)bRow[u] * N + colBase + bCol[u]);
        *(v8h*)(&Asm[0][aRow[u] * LDA + aCol[u]]) = va;
        *(v8h*)(&Bsm[0][bRow[u] * LDB + bCol[u]]) = vb;
    }
#endif
    __syncthreads();

    // ---- steady state: stage k+1 while computing k (last tile peeled) ----
    int buf = 0;
    const int kTiles = Kd / BK;
    for (int t = 0; t < kTiles - 1; ++t) {
        const int nk = (t + 1) * BK;
#if defined(USE_ASYNC_LDS)
        #pragma unroll
        for (int u = 0; u < 2; ++u) {
            async_cp16(A  + (size_t)(rowBase + aRow[u]) * Kd + nk + aCol[u],
                       &Asm[buf ^ 1][aRow[u] * LDA + aCol[u]]);
            async_cp16(Bm + (size_t)(nk + bRow[u]) * N + colBase + bCol[u],
                       &Bsm[buf ^ 1][bRow[u] * LDB + bCol[u]]);
        }
        compute(buf);
        __builtin_amdgcn_s_wait_asynccnt(0);
#else
        v8h sa[2], sb[2];
        #pragma unroll
        for (int u = 0; u < 2; ++u) {
            sa[u] = *(const v8h*)(A  + (size_t)(rowBase + aRow[u]) * Kd + nk + aCol[u]);
            sb[u] = *(const v8h*)(Bm + (size_t)(nk + bRow[u]) * N + colBase + bCol[u]);
        }
        compute(buf);
        #pragma unroll
        for (int u = 0; u < 2; ++u) {
            *(v8h*)(&Asm[buf ^ 1][aRow[u] * LDA + aCol[u]]) = sa[u];
            *(v8h*)(&Bsm[buf ^ 1][bRow[u] * LDB + bCol[u]]) = sb[u];
        }
#endif
        __syncthreads();
        buf ^= 1;
    }
    compute(buf);

    // ---- epilogue: straight-line, pointer-stepped stores ----
    const int r0 = rowBase + wm * 32 + hf * 8;
    #pragma unroll
    for (int mi = 0; mi < 2; ++mi)
        #pragma unroll
        for (int ni = 0; ni < 4; ++ni) {
            const int c = colBase + wn * 64 + ni * 16 + rw;
            const float bv = bias[c];
            size_t off = (size_t)(r0 + mi * 16) * N + c;
            #pragma unroll
            for (int i = 0; i < 8; ++i, off += N) {
                float v = acc[mi][ni][i] + bv;
                if (RELU) v = fmaxf(v, 0.f);
                if (HAS_RES) v += res[off];
                if (HAS_CF) Cf[off] = v;
                if (HAS_CH) Ch[off] = (_Float16)v;
            }
        }
}

// ----------------------------------------------------------------------------
// Depthwise conv along S (K=7, pad 3), [B,S,D] f32 in, f16 out (+bias).
// ----------------------------------------------------------------------------
__global__ void dwconv_kernel(const float* __restrict__ x, const float* __restrict__ w,
                              const float* __restrict__ bias, _Float16* __restrict__ y)
{
    int idx = blockIdx.x * blockDim.x + threadIdx.x;
    if (idx >= B_DIM * S_LEN * (D_DIM / 4)) return;
    int d4 = idx % (D_DIM / 4);
    int s  = (idx / (D_DIM / 4)) % S_LEN;
    int b  = idx / ((D_DIM / 4) * S_LEN);
    int d0 = d4 * 4;

    float wk[KW][4];
    #pragma unroll
    for (int t = 0; t < KW; ++t)
        #pragma unroll
        for (int c = 0; c < 4; ++c) wk[t][c] = w[(d0 + c) * KW + t];

    float acc[4];
    #pragma unroll
    for (int c = 0; c < 4; ++c) acc[c] = bias[d0 + c];

    #pragma unroll
    for (int t = 0; t < KW; ++t) {
        int ss = s + t - 3;
        if (ss >= 0 && ss < S_LEN) {
            v4f xv = *(const v4f*)(x + ((size_t)b * S_LEN + ss) * D_DIM + d0);
            #pragma unroll
            for (int c = 0; c < 4; ++c) acc[c] += xv[c] * wk[t][c];
        }
    }
    v4h ov;
    #pragma unroll
    for (int c = 0; c < 4; ++c) ov[c] = (_Float16)acc[c];
    *(v4h*)(y + ((size_t)b * S_LEN + s) * D_DIM + d0) = ov;
}

__global__ void cvt_f16_kernel(const float* __restrict__ src, _Float16* __restrict__ dst, int n)
{
    int i = blockIdx.x * blockDim.x + threadIdx.x;
    if (i < n) dst[i] = (_Float16)src[i];
}

// pwTH[l][in][out] = (f16) pw_w[l][out][in]
__global__ void transpose_pw_kernel(const float* __restrict__ pw, _Float16* __restrict__ pwTH)
{
    int idx = blockIdx.x * blockDim.x + threadIdx.x;
    if (idx >= 2 * D_DIM * D_DIM) return;
    int i = idx % D_DIM;
    int o = (idx / D_DIM) % D_DIM;
    int l = idx / (D_DIM * D_DIM);
    pwTH[(size_t)l * D_DIM * D_DIM + (size_t)i * D_DIM + o] =
        (_Float16)pw[(size_t)l * D_DIM * D_DIM + (size_t)o * D_DIM + i];
}

// qkv [B,S,3D] f32 -> Q f16 [BH,S,DH], K^T f16 [BH,DH,S], V f16 [BH,S,DH]
__global__ void qkv_split_kernel(const float* __restrict__ qkv, _Float16* __restrict__ Qh,
                                 _Float16* __restrict__ KTh, _Float16* __restrict__ Vh)
{
    int idx = blockIdx.x * blockDim.x + threadIdx.x;
    if (idx >= B_DIM * S_LEN * D_DIM) return;
    int d = idx % D_DIM;
    int s = (idx / D_DIM) % S_LEN;
    int b = idx / (D_DIM * S_LEN);
    int h = d >> 6, dh = d & 63;
    int bh = b * H_DIM + h;
    size_t base = ((size_t)b * S_LEN + s) * (3 * D_DIM);
    Qh [((size_t)bh * S_LEN + s) * DH_DIM + dh] = (_Float16)qkv[base + d];
    KTh[((size_t)bh * DH_DIM + dh) * S_LEN + s] = (_Float16)qkv[base + D_DIM + d];
    Vh [((size_t)bh * S_LEN + s) * DH_DIM + dh] = (_Float16)qkv[base + 2 * D_DIM + d];
}

// S2all[bh,q,r] = Q[bh,q,:] . peK[r,:]
__global__ void s2_kernel(const float* __restrict__ qkv, const float* __restrict__ peK,
                          float* __restrict__ S2all)
{
    int idx = blockIdx.x * blockDim.x + threadIdx.x;
    if (idx >= B_DIM * H_DIM * S_LEN * NR) return;
    int r  = idx % NR;
    int q  = (idx / NR) % S_LEN;
    int bh = idx / (NR * S_LEN);
    int b = bh >> 3, h = bh & 7;
    const float* qv = qkv + ((size_t)b * S_LEN + q) * (3 * D_DIM) + h * DH_DIM;
    const float* pk = peK + r * DH_DIM;
    float acc = 0.f;
    #pragma unroll
    for (int d = 0; d < DH_DIM; ++d) acc += qv[d] * pk[d];
    S2all[idx] = acc;
}

// ----------------------------------------------------------------------------
// Fused rel-pos attention: 1 wave per (bh, 16-row q-tile); f16 O output.
// Scores staged f16 in LDS, exponentiated IN PLACE after row-max so the PV
// pass is pure ds_load_b128 + WMMA (no exp/cvt in the hot loop).
// ----------------------------------------------------------------------------
__global__ __launch_bounds__(32)
void attn_kernel(const _Float16* __restrict__ Qh, const _Float16* __restrict__ KTh,
                 const _Float16* __restrict__ Vh, const float* __restrict__ S2all,
                 const int* __restrict__ mask, const float* __restrict__ peV,
                 _Float16* __restrict__ attOh)
{
    __shared__ _Float16 sc[16][S_LEN];       // 32 KB: scores, then P (in place)
    __shared__ float s2t[16][NR];            // staged S2 tile
    __shared__ float rowm[16];
    __shared__ float rowl[16];
    __shared__ float arel[16][NR];
    __shared__ float o2buf[16][DH_DIM];

    const int lane = threadIdx.x;
    const int hf = lane >> 4, rw = lane & 15;
    const int bh = blockIdx.x;
    const int b  = bh >> 3, h = bh & 7;
    const int q0 = blockIdx.y * 16;

    const _Float16* Qbase = Qh  + (size_t)bh * S_LEN * DH_DIM;
    const _Float16* Kbase = KTh + (size_t)bh * DH_DIM * S_LEN;
    const _Float16* Vbase = Vh  + (size_t)bh * S_LEN * DH_DIM;
    const float* S2base = S2all + ((size_t)bh * S_LEN + q0) * NR;
    const int* mrow = mask + b * S_LEN;

    // stage S2 tile (16x33) into LDS
    for (int t = lane; t < 16 * NR; t += 32) ((float*)s2t)[t] = S2base[t];

    v16h aq[2];
    {
        const _Float16* qrow = Qbase + (size_t)(q0 + rw) * DH_DIM;
        #pragma unroll
        for (int st = 0; st < 2; ++st) {
            v8h lo = *(const v8h*)(qrow + st * 32 + hf * 8);
            v8h hi = *(const v8h*)(qrow + st * 32 + 16 + hf * 8);
            #pragma unroll
            for (int j = 0; j < 8; ++j) { aq[st][j] = lo[j]; aq[st][8 + j] = hi[j]; }
        }
    }
    __syncthreads();

    // ---- Pass 1: scores = (QK^T + S2)/8, masked, into LDS; track row max ----
    float mx[8];
    #pragma unroll
    for (int i = 0; i < 8; ++i) mx[i] = -3.0e38f;

    for (int kt = 0; kt < S_LEN / 16; ++kt) {
        v8f c;
        #pragma unroll
        for (int e = 0; e < 8; ++e) c[e] = 0.f;
        #pragma unroll
        for (int st = 0; st < 2; ++st) {
            const _Float16* kp = Kbase + (size_t)(st * 32) * S_LEN + kt * 16 + rw;
            v16h bfv;
            #pragma unroll
            for (int j = 0; j < 8; ++j) { bfv[j] = kp[(size_t)(hf * 8 + j) * S_LEN]; bfv[8 + j] = kp[(size_t)(16 + hf * 8 + j) * S_LEN]; }
            c = __builtin_amdgcn_wmma_f32_16x16x32_f16(false, aq[st], false, bfv,
                                                       (short)0, c, false, false);
        }
        const int kk  = kt * 16 + rw;
        const int msk = mrow[kk];
        #pragma unroll
        for (int i = 0; i < 8; ++i) {
            const int m = hf * 8 + i;
            int r = kk - (q0 + m);
            r = (r < -16) ? -16 : (r > 16 ? 16 : r);
            float s = (c[i] + s2t[m][r + 16]) * 0.125f;
            if (msk == 0) s = -30000.f;
            sc[m][kk] = (_Float16)s;
            mx[i] = fmaxf(mx[i], s);
        }
    }
    #pragma unroll
    for (int i = 0; i < 8; ++i) {
        float v = mx[i];
        #pragma unroll
        for (int off = 1; off < 16; off <<= 1) v = fmaxf(v, __shfl_xor(v, off, 32));
        if (rw == 0) rowm[hf * 8 + i] = v;
    }
    __syncthreads();

    // ---- exponentiate in place + edge bins + denominator (v8h chunks) ----
    {
        const int m = rw, sub = hf;          // 2 lanes per row, 512 cols each
        const int q = q0 + m;
        const float rm = rowm[m];
        float b0p = 0.f, b32p = 0.f, lp = 0.f;
        const int c0 = sub * 512;
        for (int cc = 0; cc < 512; cc += 8) {
            const int kk = c0 + cc;
            v8h vsc = *(v8h*)(&sc[m][kk]);
            float pv[8];
            #pragma unroll
            for (int j = 0; j < 8; ++j) { pv[j] = __expf((float)vsc[j] - rm); lp += pv[j]; }
            #pragma unroll
            for (int j = 0; j < 8; ++j) {
                int r = kk + j - q;
                if (r <= -16) b0p += pv[j]; else if (r >= 16) b32p += pv[j];
            }
            v8h outv;
            #pragma unroll
            for (int j = 0; j < 8; ++j) outv[j] = (_Float16)pv[j];
            *(v8h*)(&sc[m][kk]) = outv;
        }
        b0p  += __shfl_xor(b0p, 16, 32);
        b32p += __shfl_xor(b32p, 16, 32);
        lp   += __shfl_xor(lp, 16, 32);
        if (sub == 0) { rowl[m] = lp; arel[m][0] = b0p; arel[m][32] = b32p; }
    }
    __syncthreads();

    // interior bins: single P elements, already exponentiated
    for (int t = lane; t < 16 * 31; t += 32) {
        const int m = t / 31, rr = 1 + (t % 31);
        const int kk = q0 + m + rr - 16;
        float v = 0.f;
        if (kk >= 0 && kk < S_LEN) v = (float)sc[m][kk];
        arel[m][rr] = v;
    }
    __syncthreads();

    // ---- O2 = Arel @ peV ----
    {
        const int m = rw;
        const int dbase = hf * 32;
        for (int cix = 0; cix < 32; ++cix) {
            float a2 = 0.f;
            #pragma unroll
            for (int r = 0; r < NR; ++r) a2 += arel[m][r] * peV[r * DH_DIM + dbase + cix];
            o2buf[m][dbase + cix] = a2;
        }
    }
    __syncthreads();

    // ---- Pass 2: O1 = P @ V, pure ds_load + WMMA ----
    v8f oacc[4];
    #pragma unroll
    for (int t = 0; t < 4; ++t)
        #pragma unroll
        for (int e = 0; e < 8; ++e) oacc[t][e] = 0.f;

    for (int kc = 0; kc < S_LEN / 32; ++kc) {
        v16h pa;
        v8h plo = *(const v8h*)(&sc[rw][kc * 32 + hf * 8]);
        v8h phi = *(const v8h*)(&sc[rw][kc * 32 + 16 + hf * 8]);
        #pragma unroll
        for (int j = 0; j < 8; ++j) { pa[j] = plo[j]; pa[8 + j] = phi[j]; }
        const _Float16* vrow = Vbase + (size_t)(kc * 32) * DH_DIM + rw;
        #pragma unroll
        for (int t = 0; t < 4; ++t) {
            v16h vb;
            #pragma unroll
            for (int j = 0; j < 8; ++j) { vb[j] = vrow[(hf * 8 + j) * DH_DIM + t * 16]; vb[8 + j] = vrow[(16 + hf * 8 + j) * DH_DIM + t * 16]; }
            oacc[t] = __builtin_amdgcn_wmma_f32_16x16x32_f16(false, pa, false, vb,
                                                             (short)0, oacc[t], false, false);
        }
    }

    #pragma unroll
    for (int t = 0; t < 4; ++t)
        #pragma unroll
        for (int i = 0; i < 8; ++i) {
            const int m  = hf * 8 + i;
            const int dh = t * 16 + rw;
            const float val = (oacc[t][i] + o2buf[m][dh]) / rowl[m];
            attOh[((size_t)b * S_LEN + q0 + m) * D_DIM + h * DH_DIM + dh] = (_Float16)val;
        }
}

// ----------------------------------------------------------------------------
extern "C" void kernel_launch(void* const* d_in, const int* in_sizes, int n_in,
                              void* d_out, int out_size, void* d_ws, size_t ws_size,
                              hipStream_t stream)
{
    const float* x    = (const float*)d_in[0];
    const int*   mask = (const int*)  d_in[1];
    const float* dw_w = (const float*)d_in[2];
    const float* dw_b = (const float*)d_in[3];
    const float* pw_w = (const float*)d_in[4];
    const float* pw_b = (const float*)d_in[5];
    const float* Wqkv = (const float*)d_in[6];
    const float* bqkv = (const float*)d_in[7];
    const float* Wo   = (const float*)d_in[8];
    const float* bo   = (const float*)d_in[9];
    const float* peK  = (const float*)d_in[10];
    const float* peV  = (const float*)d_in[11];
    const float* W1   = (const float*)d_in[12];
    const float* b1   = (const float*)d_in[13];
    const float* W2   = (const float*)d_in[14];
    const float* b2   = (const float*)d_in[15];
    float* out = (float*)d_out;

    // ---- workspace layout (~151 MB) ----
    const size_t NTOK = (size_t)B_DIM * S_LEN;           // 8192
    const size_t ND   = NTOK * D_DIM;                    // 4,194,304
    float* wsf  = (float*)d_ws;
    float* bufX = wsf;                                   // f32 [NTOK,D]
    float* qkvB = bufX + ND;                             // f32 [NTOK,3D]
    float* attB = qkvB + 3 * ND;                         // f32 [NTOK,D]
    float* S2a  = attB + ND;                             // f32 [64,S,33]
    _Float16* dwh   = (_Float16*)(S2a + (size_t)B_DIM * H_DIM * S_LEN * NR);
    _Float16* bufXh = dwh   + ND;
    _Float16* Qh    = bufXh + ND;
    _Float16* KTh   = Qh    + ND;
    _Float16* Vh    = KTh   + ND;
    _Float16* attBh = Vh    + ND;
    _Float16* WqkvH = attBh + ND;
    _Float16* WoH   = WqkvH + (size_t)D_DIM * 3 * D_DIM;
    _Float16* W1H   = WoH   + (size_t)D_DIM * D_DIM;
    _Float16* W2H   = W1H   + (size_t)D_DIM * DFF_DIM;
    _Float16* pwTH  = W2H   + (size_t)DFF_DIM * D_DIM;
    _Float16* attOh = dwh;               // alias: dwh dead after conv stack
    _Float16* hh    = (_Float16*)qkvB;   // alias: qkv dead after split/s2

    const dim3 gblk(32, 8);

    // ---- convert weights once ----
    cvt_f16_kernel<<<(3*D_DIM*D_DIM/256)+1, 256, 0, stream>>>(Wqkv, WqkvH, 3*D_DIM*D_DIM);
    cvt_f16_kernel<<<(D_DIM*D_DIM/256)+1, 256, 0, stream>>>(Wo, WoH, D_DIM*D_DIM);
    cvt_f16_kernel<<<(D_DIM*DFF_DIM/256)+1, 256, 0, stream>>>(W1, W1H, D_DIM*DFF_DIM);
    cvt_f16_kernel<<<(DFF_DIM*D_DIM/256)+1, 256, 0, stream>>>(W2, W2H, DFF_DIM*D_DIM);
    transpose_pw_kernel<<<(2*D_DIM*D_DIM + 255)/256, 256, 0, stream>>>(pw_w, pwTH);

    // ---- conv stack ----
    dwconv_kernel<<<(B_DIM*S_LEN*(D_DIM/4) + 255)/256, 256, 0, stream>>>(x, dw_w, dw_b, dwh);
    wmma_gemm_f16<1, true, true, true><<<dim3(NTOK/BM, D_DIM/BN), gblk, 0, stream>>>(
        dwh, pwTH, pw_b, x, bufX, bufXh, (int)NTOK, D_DIM, D_DIM);

    dwconv_kernel<<<(B_DIM*S_LEN*(D_DIM/4) + 255)/256, 256, 0, stream>>>(
        bufX, dw_w + D_DIM*KW, dw_b + D_DIM, dwh);
    wmma_gemm_f16<1, true, true, true><<<dim3(NTOK/BM, D_DIM/BN), gblk, 0, stream>>>(
        dwh, pwTH + D_DIM*D_DIM, pw_b + D_DIM, bufX, bufX, bufXh, (int)NTOK, D_DIM, D_DIM);

    // ---- attention ----
    wmma_gemm_f16<0, false, true, false><<<dim3(NTOK/BM, (3*D_DIM)/BN), gblk, 0, stream>>>(
        bufXh, WqkvH, bqkv, nullptr, qkvB, nullptr, (int)NTOK, 3*D_DIM, D_DIM);
    qkv_split_kernel<<<(B_DIM*S_LEN*D_DIM + 255)/256, 256, 0, stream>>>(qkvB, Qh, KTh, Vh);
    s2_kernel<<<(B_DIM*H_DIM*S_LEN*NR + 255)/256, 256, 0, stream>>>(qkvB, peK, S2a);
    attn_kernel<<<dim3(B_DIM*H_DIM, S_LEN/16), dim3(32), 0, stream>>>(
        Qh, KTh, Vh, S2a, mask, peV, attOh);
    wmma_gemm_f16<0, true, true, true><<<dim3(NTOK/BM, D_DIM/BN), gblk, 0, stream>>>(
        attOh, WoH, bo, bufX, attB, attBh, (int)NTOK, D_DIM, D_DIM);

    // ---- FFN ----
    wmma_gemm_f16<1, false, false, true><<<dim3(NTOK/BM, DFF_DIM/BN), gblk, 0, stream>>>(
        attBh, W1H, b1, nullptr, nullptr, hh, (int)NTOK, DFF_DIM, D_DIM);
    wmma_gemm_f16<1, true, true, false><<<dim3(NTOK/BM, D_DIM/BN), gblk, 0, stream>>>(
        hh, W2H, b2, attB, out, nullptr, (int)NTOK, D_DIM, DFF_DIM);
}
